// EpistemicMemory_7919919694526
// MI455X (gfx1250) — compile-verified, matrix operand order checked by reading, not run on hardware
//
#include <hip/hip_runtime.h>
#include <math.h>

// ---------------------------------------------------------------------------
// EpistemicMemory retrieval for MI455X (gfx1250, wave32, WMMA)
// v3: 4 M-subtiles x 2 N-tiles per wave -> each B fragment feeds 4 WMMAs,
//     cutting L2 key traffic to ~2 GB while keeping 8 accumulators/wave.
// ---------------------------------------------------------------------------

typedef _Float16 v4h  __attribute__((ext_vector_type(4)));
typedef _Float16 v8h  __attribute__((ext_vector_type(8)));
typedef _Float16 v16h __attribute__((ext_vector_type(16)));
typedef float    v8f  __attribute__((ext_vector_type(8)));

#define BT        2048          // B*T query rows
#define DK        512
#define DV        512
#define NKEYS     65536
#define KTOP      32
#define INV_TEMP  10.0f         // 1 / 0.1

#define QTILE     64            // query rows per block (4 WMMA M-subtiles)
#define CHUNK     256           // keys per block
#define NCHUNKS   (NKEYS / CHUNK)   // 256
#define QTILES    (BT / QTILE)      // 32
#define NCAND     (NCHUNKS * KTOP)  // 8192 candidates per row at merge

#define QSTRIDE   (DK + 8)      // padded LDS row stride (halves) -> conflict-free b128
#define SSTRIDE   (CHUNK + 4)   // padded LDS score stride (floats)

// ---------------------------------------------------------------------------
// Kernel 1: row-wise L2 normalize f32 -> f16  (x / max(||x||, 1e-12))
// ---------------------------------------------------------------------------
__global__ __launch_bounds__(128) void normalize_rows_f16(
    const float* __restrict__ src, _Float16* __restrict__ dst) {
  const int row = blockIdx.x;
  const int t   = threadIdx.x;
  const float4 x = ((const float4*)(src + (size_t)row * DK))[t];
  float ss = x.x * x.x + x.y * x.y + x.z * x.z + x.w * x.w;
#pragma unroll
  for (int m = 16; m >= 1; m >>= 1) ss += __shfl_xor(ss, m);
  __shared__ float wsum[4];
  if ((t & 31) == 0) wsum[t >> 5] = ss;
  __syncthreads();
  const float tot   = wsum[0] + wsum[1] + wsum[2] + wsum[3];
  const float scale = 1.0f / fmaxf(sqrtf(tot), 1e-12f);
  v4h o;
  o[0] = (_Float16)(x.x * scale);
  o[1] = (_Float16)(x.y * scale);
  o[2] = (_Float16)(x.z * scale);
  o[3] = (_Float16)(x.w * scale);
  ((v4h*)(dst + (size_t)row * DK))[t] = o;
}

// ---------------------------------------------------------------------------
// Kernel 2: WMMA score tile + per-chunk top-32
// grid = (QTILES, NCHUNKS); block = 256 threads = 8 waves.
// Each wave: 64 (M) x 32 (N) via 8 accumulators; K=512 by 16x16x32 f16 WMMA.
// ---------------------------------------------------------------------------
__global__ __launch_bounds__(256) void score_topk_chunk(
    const _Float16* __restrict__ qn, const _Float16* __restrict__ kn,
    float* __restrict__ cand_s, int* __restrict__ cand_i) {
  __shared__ _Float16 s_q[QTILE * QSTRIDE];   // 66.5 KB
  __shared__ float    s_sc[QTILE * SSTRIDE];  // 66.5 KB

  const int qtile = blockIdx.x;
  const int chunk = blockIdx.y;
  const int tid   = threadIdx.x;

  // stage 64x512 f16 query tile into padded LDS (4096 uint4 total)
  {
    const uint4* g = (const uint4*)(qn + (size_t)qtile * QTILE * DK);
#pragma unroll
    for (int i = 0; i < 16; ++i) {
      const int idx = tid + 256 * i;   // = r*64 + c
      const int r = idx >> 6;
      const int c = idx & 63;
      *(uint4*)(s_q + r * QSTRIDE + c * 8) = g[idx];
    }
  }
  __syncthreads();

  const int wave = tid >> 5;
  const int lane = tid & 31;
  const int h16  = lane & 15;
  const int sel  = lane >> 4;                  // K half-group this lane holds
  const int colbase = wave * (CHUNK / 8);      // 32-key column strip per wave

  const _Float16* arow[4];
#pragma unroll
  for (int m = 0; m < 4; ++m) arow[m] = s_q + (m * 16 + h16) * QSTRIDE;

  const _Float16* brow[2];
#pragma unroll
  for (int nt = 0; nt < 2; ++nt) {
    brow[nt] = kn + (size_t)(chunk * CHUNK + colbase + nt * 16 + h16) * DK;
    __builtin_prefetch((const void*)brow[nt], 0, 1);
  }

  v8f acc[8];  // [nt*4 + m]
#pragma unroll
  for (int i = 0; i < 8; ++i) acc[i] = (v8f){};

#pragma unroll
  for (int kt = 0; kt < DK / 32; ++kt) {       // 16 K-steps of 32
    const int kb = kt * 32 + sel * 8;
    // A fragments: 4 M-subtiles, loaded once per k-step
    v16h a[4];
#pragma unroll
    for (int m = 0; m < 4; ++m) {
      const v8h al = *(const v8h*)(arow[m] + kb);
      const v8h ah = *(const v8h*)(arow[m] + kb + 16);
      a[m] = __builtin_shufflevector(al, ah, 0,1,2,3,4,5,6,7,8,9,10,11,12,13,14,15);
    }
#pragma unroll
    for (int nt = 0; nt < 2; ++nt) {
      const v8h bl = *(const v8h*)(brow[nt] + kb);
      const v8h bh = *(const v8h*)(brow[nt] + kb + 16);
      const v16h b = __builtin_shufflevector(bl, bh, 0,1,2,3,4,5,6,7,8,9,10,11,12,13,14,15);
#pragma unroll
      for (int m = 0; m < 4; ++m)
        acc[nt * 4 + m] = __builtin_amdgcn_wmma_f32_16x16x32_f16(
            false, a[m], false, b, (short)0, acc[nt * 4 + m], false, false);
    }
  }

  // C/D layout: vgpr r, lane l -> score[m*16 + r + 8*(l>>4)][ncol + (l&15)]
#pragma unroll
  for (int nt = 0; nt < 2; ++nt) {
    const int ncol = colbase + nt * 16 + h16;
#pragma unroll
    for (int m = 0; m < 4; ++m)
#pragma unroll
      for (int r = 0; r < 8; ++r)
        s_sc[(m * 16 + r + 8 * sel) * SSTRIDE + ncol] = acc[nt * 4 + m][r];
  }
  __syncthreads();

  // per-row chunk-local top-32 (wave w owns rows 8w..8w+7); stable tie-break
  for (int rr = 0; rr < QTILE / 8; ++rr) {
    const int row = wave * (QTILE / 8) + rr;
    const size_t obase =
        ((size_t)(qtile * QTILE + row) * NCHUNKS + chunk) * KTOP;
    for (int it = 0; it < KTOP; ++it) {
      float bs = -__builtin_inff();
      int   bc = 0x7fffffff;
#pragma unroll
      for (int j = 0; j < CHUNK / 32; ++j) {
        const int c = lane + 32 * j;
        const float s = s_sc[row * SSTRIDE + c];
        if (s > bs) { bs = s; bc = c; }   // first hit => smallest c per lane
      }
#pragma unroll
      for (int m = 16; m >= 1; m >>= 1) {
        const float os = __shfl_xor(bs, m);
        const int   oc = __shfl_xor(bc, m);
        if (os > bs || (os == bs && oc < bc)) { bs = os; bc = oc; }
      }
      if (lane == it) {
        cand_s[obase + it] = bs;
        cand_i[obase + it] = chunk * CHUNK + bc;
      }
      if (lane == 0) s_sc[row * SSTRIDE + bc] = -__builtin_inff();
      __builtin_amdgcn_wave_barrier();   // keep LDS invalidate ordered vs rescan
    }
  }
}

// ---------------------------------------------------------------------------
// Kernel 3: merge 256 chunk-lists -> global top-32, softmax, value aggregate
// one block (256 threads) per query row
// ---------------------------------------------------------------------------
__global__ __launch_bounds__(256) void merge_topk_aggregate(
    const float* __restrict__ cand_s, const int* __restrict__ cand_i,
    const float* __restrict__ values, float* __restrict__ out_agg,
    float* __restrict__ out_attn, float* __restrict__ out_idx) {
  __shared__ float s_sc[NCAND];
  __shared__ int   s_ix[NCAND];
  __shared__ float s_attn[KTOP];
  __shared__ int   s_tix[KTOP];

  const int row = blockIdx.x;
  const int tid = threadIdx.x;
  {
    const float* cs = cand_s + (size_t)row * NCAND;
    const int*   ci = cand_i + (size_t)row * NCAND;
#pragma unroll
    for (int i = 0; i < NCAND / 256; ++i) {
      s_sc[tid + 256 * i] = cs[tid + 256 * i];
      s_ix[tid + 256 * i] = ci[tid + 256 * i];
    }
  }
  __syncthreads();

  if (tid < 32) {
    const int lane = tid;
    float sel_s = 0.0f;
    int   sel_i = 0;
    for (int it = 0; it < KTOP; ++it) {
      float bs = -__builtin_inff();
      int   bi = 0x7fffffff;
      int   bslot = 0;
      for (int j = 0; j < NCAND / 32; ++j) {
        const int slot = lane + 32 * j;
        const float s = s_sc[slot];
        const int  ix = s_ix[slot];
        if (s > bs || (s == bs && ix < bi)) { bs = s; bi = ix; bslot = slot; }
      }
#pragma unroll
      for (int m = 16; m >= 1; m >>= 1) {
        const float os  = __shfl_xor(bs, m);
        const int   oi  = __shfl_xor(bi, m);
        const int   osl = __shfl_xor(bslot, m);
        if (os > bs || (os == bs && oi < bi)) { bs = os; bi = oi; bslot = osl; }
      }
      if (lane == 0) s_sc[bslot] = -__builtin_inff();
      __builtin_amdgcn_wave_barrier();
      if (lane == it) { sel_s = bs; sel_i = bi; }
    }
    // softmax over the 32 selections (descending -> lane 0 holds the max)
    const float m0 = __shfl(sel_s, 0);
    const float e  = __expf((sel_s - m0) * INV_TEMP);
    float sum = e;
#pragma unroll
    for (int m = 16; m >= 1; m >>= 1) sum += __shfl_xor(sum, m);
    const float a = e / sum;
    s_attn[lane] = a;
    s_tix[lane]  = sel_i;
    out_attn[(size_t)row * KTOP + lane] = a;
    out_idx[(size_t)row * KTOP + lane]  = (float)sel_i;  // < 2^24, exact in f32
  }
  __syncthreads();

  // weighted gather of value rows (f32 exact); coalesced 2 dims per thread
  float a0 = 0.0f, a1 = 0.0f;
#pragma unroll 4
  for (int j = 0; j < KTOP; ++j) {
    const float w = s_attn[j];
    const float* vr = values + (size_t)s_tix[j] * DV;
    a0 += w * vr[tid];
    a1 += w * vr[tid + 256];
  }
  out_agg[(size_t)row * DV + tid]       = a0;
  out_agg[(size_t)row * DV + tid + 256] = a1;
}

// ---------------------------------------------------------------------------
extern "C" void kernel_launch(void* const* d_in, const int* in_sizes, int n_in,
                              void* d_out, int out_size, void* d_ws, size_t ws_size,
                              hipStream_t stream) {
  (void)in_sizes; (void)n_in; (void)out_size; (void)ws_size;
  const float* q = (const float*)d_in[0];   // [4,512,512]
  const float* k = (const float*)d_in[1];   // [65536,512]
  const float* v = (const float*)d_in[2];   // [65536,512]

  float* out      = (float*)d_out;
  float* out_agg  = out;                               // 2048*512
  float* out_attn = out + (size_t)BT * DV;             // 2048*32
  float* out_idx  = out_attn + (size_t)BT * KTOP;      // 2048*32

  char* ws = (char*)d_ws;
  _Float16* qn = (_Float16*)ws;                                       // 2 MB
  _Float16* kn = (_Float16*)(ws + (size_t)BT * DK * 2);               // 64 MB
  float* cand_s = (float*)(ws + (size_t)BT * DK * 2 + (size_t)NKEYS * DK * 2);  // 64 MB
  int*   cand_i = (int*)((char*)cand_s + (size_t)BT * NCHUNKS * KTOP * sizeof(float));

  normalize_rows_f16<<<BT,    128, 0, stream>>>(q, qn);
  normalize_rows_f16<<<NKEYS, 128, 0, stream>>>(k, kn);
  score_topk_chunk<<<dim3(QTILES, NCHUNKS), 256, 0, stream>>>(qn, kn, cand_s, cand_i);
  merge_topk_aggregate<<<BT, 256, 0, stream>>>(cand_s, cand_i, v,
                                               out_agg, out_attn, out_idx);
}
